// Attention_43310450212960
// MI455X (gfx1250) — compile-verified, matrix operand order checked by reading, not run on hardware
//
#include <hip/hip_runtime.h>

typedef float  v8f  __attribute__((ext_vector_type(8)));
typedef __bf16 v16bf __attribute__((ext_vector_type(16)));
typedef unsigned int u32x4 __attribute__((ext_vector_type(4)));
typedef int          i32x8 __attribute__((ext_vector_type(8)));
typedef int          i32x4 __attribute__((ext_vector_type(4)));

union Frag {
  v16bf v;
  uint4 u[2];
};

__device__ __forceinline__ unsigned short f2bf(float f) {
  unsigned int u = __builtin_bit_cast(unsigned int, f);
  u += 0x7FFFu + ((u >> 16) & 1u);   // round-to-nearest-even
  return (unsigned short)(u >> 16);
}

// ---------------------------------------------------------------------------
// TDM: 2D bf16 tile load global -> LDS with optional per-row LDS padding.
// D# layout per CDNA5 ISA ch.8 (group0: ctrl/lds/global/type, group1: dims).
// pad_interval: 0=2,1=4,2=8,3=16,4=32,... DWORDs; pad_amount: N+1 DWORDs.
// This toolchain exposes the 6-arg builtin (g0, g1, g2, g3, g1ext, cpol).
// ---------------------------------------------------------------------------
__device__ __forceinline__ void tdm_load_2d_bf16(
    unsigned lds_addr, const void* gptr,
    unsigned tensor_d0, unsigned tensor_d1,
    unsigned tile_d0, unsigned tile_d1,
    unsigned long long stride0_elems,
    int pad_en, unsigned pad_interval, unsigned pad_amount) {
  unsigned long long ga = (unsigned long long)gptr;
  u32x4 g0;
  g0[0] = 1u;                                   // count=1, user descriptor
  g0[1] = lds_addr;                             // LDS byte address
  g0[2] = (unsigned)(ga & 0xFFFFFFFFu);         // global_addr[31:0]
  g0[3] = (unsigned)((ga >> 32) & 0x1FFFFFFu)   // global_addr[56:32]
          | (2u << 30);                         // type = 2 ("image")
  i32x8 g1;
  unsigned w0 = (1u << 16);                     // data_size = 1 -> 2 bytes
  if (pad_en) w0 |= (1u << 20) | (pad_interval << 22) | (pad_amount << 25);
  g1[0] = (int)w0;
  g1[1] = (int)((tensor_d0 & 0xFFFFu) << 16);                 // dim0[15:0]
  g1[2] = (int)((tensor_d0 >> 16) | ((tensor_d1 & 0xFFFFu) << 16));
  g1[3] = (int)((tensor_d1 >> 16) | (tile_d0 << 16));
  g1[4] = (int)tile_d1;                                       // tile_dim2 = 0
  g1[5] = (int)(unsigned)(stride0_elems & 0xFFFFFFFFull);
  g1[6] = (int)(unsigned)((stride0_elems >> 32) & 0xFFFFull); // stride1 = 0
  g1[7] = 0;
  i32x4 z4 = {0, 0, 0, 0};
  i32x8 z8 = {0, 0, 0, 0, 0, 0, 0, 0};
  __builtin_amdgcn_tensor_load_to_lds(g0, g1, z4, z4, z8, 0);
}

constexpr int   Bb = 4, Nn = 2048, Cc = 512, Hh = 8, Dd = 64;
constexpr int   M  = Bb * Nn;              // 8192 rows
constexpr float SCALE = 0.125f;            // 64^-0.5

// ---------------------------------------------------------------------------
// fp32 -> bf16 conversion (grid-stride)
// ---------------------------------------------------------------------------
__global__ void f32_to_bf16_kernel(const float* __restrict__ src,
                                   unsigned short* __restrict__ dst, int n) {
  for (int i = blockIdx.x * blockDim.x + threadIdx.x; i < n;
       i += gridDim.x * blockDim.x)
    dst[i] = f2bf(src[i]);
}

// ---------------------------------------------------------------------------
// bf16 GEMM: C[m,f] = sum_k A[m,k] * W[f,k] + bias[f]
// MODE 0: scatter into q/k/v [B,H,N,d] bf16 (q pre-scaled by SCALE)
// MODE 1: fp32 row-major output
// Block = 256 thr (8 waves); tile 128(M) x 128(N); K-chunk 32 staged by TDM.
// ---------------------------------------------------------------------------
template <int MODE>
__global__ __launch_bounds__(256) void gemm_bf16_kernel(
    const unsigned short* __restrict__ A,   // [M, 512] bf16
    const unsigned short* __restrict__ W,   // [NF, 512] bf16
    const float* __restrict__ bias,         // [NF]
    unsigned short* __restrict__ qout,
    unsigned short* __restrict__ kout,
    unsigned short* __restrict__ vout,
    float* __restrict__ fout) {
  __shared__ unsigned short Asm[128 * 40];  // rows padded to 80B (bank spread)
  __shared__ unsigned short Bsm[128 * 40];

  const int tid  = threadIdx.x;
  const int lane = tid & 31;
  const int wave = tid >> 5;
  const int L15  = lane & 15;
  const int g    = lane >> 4;
  const int m0   = blockIdx.x * 128;
  const int n0   = blockIdx.y * 128;
  const int waveM = wave >> 2;              // 0..1  -> 64 rows each
  const int waveN = wave & 3;               // 0..3  -> 32 cols each
  const unsigned AsmAddr = (unsigned)(size_t)(void*)Asm;
  const unsigned BsmAddr = (unsigned)(size_t)(void*)Bsm;

  const v8f vzero = {0.f, 0.f, 0.f, 0.f, 0.f, 0.f, 0.f, 0.f};
  v8f acc[4][2];
#pragma unroll
  for (int mf = 0; mf < 4; ++mf)
#pragma unroll
    for (int nf = 0; nf < 2; ++nf) acc[mf][nf] = vzero;

  for (int kk = 0; kk < Cc; kk += 32) {
    __syncthreads();
    // TDM-stage 128x32 bf16 tiles of A and W; pad 16 DW rows by 4 DW -> 80B
    if (wave == 0) {
      tdm_load_2d_bf16(AsmAddr, A + (size_t)m0 * Cc + kk,
                       Cc, 128, 32, 128, Cc, 1, 3u, 3u);
      tdm_load_2d_bf16(BsmAddr, W + (size_t)n0 * Cc + kk,
                       Cc, 128, 32, 128, Cc, 1, 3u, 3u);
      __builtin_amdgcn_s_wait_tensorcnt(0);
    }
    __syncthreads();

    Frag af[4], bfr[2];
#pragma unroll
    for (int mf = 0; mf < 4; ++mf) {
      int base = (waveM * 64 + mf * 16 + L15) * 40 + g * 8;
      af[mf].u[0] = *(const uint4*)&Asm[base];
      af[mf].u[1] = *(const uint4*)&Asm[base + 16];
    }
#pragma unroll
    for (int nf = 0; nf < 2; ++nf) {
      int base = (waveN * 32 + nf * 16 + L15) * 40 + g * 8;
      bfr[nf].u[0] = *(const uint4*)&Bsm[base];
      bfr[nf].u[1] = *(const uint4*)&Bsm[base + 16];
    }
#pragma unroll
    for (int mf = 0; mf < 4; ++mf)
#pragma unroll
      for (int nf = 0; nf < 2; ++nf)
        acc[mf][nf] = __builtin_amdgcn_wmma_f32_16x16x32_bf16(
            false, af[mf].v, false, bfr[nf].v, (short)0, acc[mf][nf],
            false, false);
  }

  // epilogue (D layout: lane holds col L15; vgpr r holds row r + 8*g)
#pragma unroll
  for (int mf = 0; mf < 4; ++mf)
#pragma unroll
    for (int nf = 0; nf < 2; ++nf) {
      int fcol  = n0 + waveN * 32 + nf * 16 + L15;
      float bv  = bias[fcol];
#pragma unroll
      for (int r = 0; r < 8; ++r) {
        int m     = m0 + waveM * 64 + mf * 16 + r + 8 * g;
        float val = acc[mf][nf][r] + bv;
        if (MODE == 0) {
          int which = fcol >> 9;            // 0=q 1=k 2=v
          int c     = fcol & 511;
          int head  = c >> 6;
          int dd    = c & 63;
          int b     = m >> 11;
          int nn2   = m & 2047;
          size_t idx = ((size_t)(b * Hh + head) * Nn + nn2) * Dd + dd;
          unsigned short obf = f2bf(which == 0 ? val * SCALE : val);
          if (which == 0)      qout[idx] = obf;
          else if (which == 1) kout[idx] = obf;
          else                 vout[idx] = obf;
        } else {
          fout[(size_t)m * Cc + fcol] = val;
        }
      }
    }
}

// ---------------------------------------------------------------------------
// Flash attention: 1 block = 4 waves = 64 queries of one (b,h).
// K tile via TDM (padded rows), V transposed manually (overlapped), softmax
// stats in registers, S/P never leave the WGP.
// ---------------------------------------------------------------------------
__global__ __launch_bounds__(128) void flash_attn_kernel(
    const unsigned short* __restrict__ q,
    const unsigned short* __restrict__ k,
    const unsigned short* __restrict__ v,
    unsigned short* __restrict__ o) {
  __shared__ unsigned short Kt[32 * 72];    // [key][d], rows padded to 144B
  __shared__ unsigned short Vt[64 * 40];    // transposed: [d][key], pad 80B
  __shared__ unsigned short Pt[4][16 * 40]; // per-wave P transpose buffer

  const int tid  = threadIdx.x;
  const int lane = tid & 31;
  const int wave = tid >> 5;
  const int L15  = lane & 15;
  const int g    = lane >> 4;
  const int bh   = blockIdx.y;
  const int b    = bh >> 3;
  const int h    = bh & 7;
  const int q0   = blockIdx.x * 64;
  const size_t base = (size_t)bh * Nn * Dd;
  const unsigned KtAddr = (unsigned)(size_t)(void*)Kt;

  // Q fragments (A layout), two 16x32 chunks across d=64; held for whole loop
  const int qrow = q0 + wave * 16 + L15;
  Frag qa[2];
#pragma unroll
  for (int dc = 0; dc < 2; ++dc) {
    size_t a = base + (size_t)qrow * Dd + dc * 32 + g * 8;
    qa[dc].u[0] = *(const uint4*)&q[a];
    qa[dc].u[1] = *(const uint4*)&q[a + 16];
  }

  const v8f vzero = {0.f, 0.f, 0.f, 0.f, 0.f, 0.f, 0.f, 0.f};
  float mstat[8], lstat[8];
  v8f oacc[4];
#pragma unroll
  for (int r = 0; r < 8; ++r) { mstat[r] = -1e30f; lstat[r] = 0.f; }
#pragma unroll
  for (int t = 0; t < 4; ++t) oacc[t] = vzero;

  for (int kb = 0; kb < Nn / 32; ++kb) {
    __syncthreads();
    size_t kbase = base + (size_t)kb * 32 * Dd;
    // K tile via TDM: 32 rows x 64 elems, 32-DW rows padded by 4 DW -> 144B
    if (wave == 0)
      tdm_load_2d_bf16(KtAddr, k + kbase, Dd, 32, Dd, 32, Dd, 1, 4u, 3u);
    // V tile transposed manually, overlapped with the TDM
#pragma unroll
    for (int i = 0; i < 2; ++i) {
      int e   = tid + i * 128;              // 256 chunks of 8 bf16
      int key = e >> 3;
      int c0  = (e & 7) * 8;
      union { uint4 u; unsigned short s[8]; } tv;
      tv.u = *(const uint4*)&v[kbase + key * 64 + c0];
#pragma unroll
      for (int j = 0; j < 8; ++j) Vt[(c0 + j) * 40 + key] = tv.s[j];
    }
    if (wave == 0) __builtin_amdgcn_s_wait_tensorcnt(0);
    __syncthreads();

    // S = Q * K^T  (two 16x16 key tiles, reduce over d=64 -> 2 WMMA each)
    v8f st[2];
#pragma unroll
    for (int kc = 0; kc < 2; ++kc) {
      v8f s = vzero;
#pragma unroll
      for (int dc = 0; dc < 2; ++dc) {
        Frag kf;
        int off = (kc * 16 + L15) * 72 + dc * 32 + g * 8;
        kf.u[0] = *(const uint4*)&Kt[off];
        kf.u[1] = *(const uint4*)&Kt[off + 16];
        s = __builtin_amdgcn_wmma_f32_16x16x32_bf16(
            false, qa[dc].v, false, kf.v, (short)0, s, false, false);
      }
      st[kc] = s;
    }

    // online softmax: per-lane stats for row r + 8*g, columns across 16 lanes
    float alpha[8];
#pragma unroll
    for (int r = 0; r < 8; ++r) {
      float mx = fmaxf(st[0][r], st[1][r]);
#pragma unroll
      for (int off = 1; off < 16; off <<= 1)
        mx = fmaxf(mx, __shfl_xor(mx, off, 32));
      float mnew = fmaxf(mstat[r], mx);
      alpha[r]   = __expf(mstat[r] - mnew);
      mstat[r]   = mnew;
      float p0 = __expf(st[0][r] - mnew);
      float p1 = __expf(st[1][r] - mnew);
      st[0][r] = p0;
      st[1][r] = p1;
      float rs = p0 + p1;
#pragma unroll
      for (int off = 1; off < 16; off <<= 1) rs += __shfl_xor(rs, off, 32);
      lstat[r] = lstat[r] * alpha[r] + rs;
    }
#pragma unroll
    for (int t = 0; t < 4; ++t)
#pragma unroll
      for (int r = 0; r < 8; ++r) oacc[t][r] *= alpha[r];

    // P: D layout f32 -> A layout bf16 via per-wave LDS round trip
    unsigned short* P = &Pt[wave][0];
#pragma unroll
    for (int kc = 0; kc < 2; ++kc)
#pragma unroll
      for (int r = 0; r < 8; ++r)
        P[(r + 8 * g) * 40 + kc * 16 + L15] = f2bf(st[kc][r]);
    __syncthreads();

    Frag pf;
    int poff = L15 * 40 + g * 8;
    pf.u[0] = *(const uint4*)&P[poff];
    pf.u[1] = *(const uint4*)&P[poff + 16];

    // O += P * V (K-dim = 32 keys, 4 d-tiles)
#pragma unroll
    for (int t = 0; t < 4; ++t) {
      Frag vf;
      int off = (t * 16 + L15) * 40 + g * 8;
      vf.u[0] = *(const uint4*)&Vt[off];
      vf.u[1] = *(const uint4*)&Vt[off + 16];
      oacc[t] = __builtin_amdgcn_wmma_f32_16x16x32_bf16(
          false, pf.v, false, vf.v, (short)0, oacc[t], false, false);
    }
  }

  // finalize: divide by l, write [B,N,C] bf16 with c = h*64 + d
#pragma unroll
  for (int t = 0; t < 4; ++t)
#pragma unroll
    for (int r = 0; r < 8; ++r) {
      int row    = q0 + wave * 16 + r + 8 * g;
      float linv = lstat[r] > 0.f ? 1.f / lstat[r] : 0.f;
      float val  = oacc[t][r] * linv;
      int d      = t * 16 + L15;
      o[((size_t)(b * Nn + row)) * Cc + h * 64 + d] = f2bf(val);
    }
}

// ---------------------------------------------------------------------------
extern "C" void kernel_launch(void* const* d_in, const int* in_sizes, int n_in,
                              void* d_out, int out_size, void* d_ws,
                              size_t ws_size, hipStream_t stream) {
  (void)in_sizes; (void)n_in; (void)out_size; (void)ws_size;
  const float* x      = (const float*)d_in[0];
  const float* w_qkv  = (const float*)d_in[1];
  const float* b_qkv  = (const float*)d_in[2];
  const float* w_proj = (const float*)d_in[3];
  const float* b_proj = (const float*)d_in[4];

  unsigned short* ws     = (unsigned short*)d_ws;
  unsigned short* xb     = ws;
  unsigned short* wqkvb  = xb + (size_t)M * Cc;
  unsigned short* wprojb = wqkvb + (size_t)3 * Cc * Cc;
  unsigned short* qb     = wprojb + (size_t)Cc * Cc;
  unsigned short* kbuf   = qb + (size_t)M * Cc;
  unsigned short* vbuf   = kbuf + (size_t)M * Cc;
  unsigned short* attb   = vbuf + (size_t)M * Cc;

  f32_to_bf16_kernel<<<512, 256, 0, stream>>>(x, xb, M * Cc);
  f32_to_bf16_kernel<<<256, 256, 0, stream>>>(w_qkv, wqkvb, 3 * Cc * Cc);
  f32_to_bf16_kernel<<<128, 256, 0, stream>>>(w_proj, wprojb, Cc * Cc);

  gemm_bf16_kernel<0><<<dim3(M / 128, (3 * Cc) / 128), 256, 0, stream>>>(
      xb, wqkvb, b_qkv, qb, kbuf, vbuf, nullptr);

  flash_attn_kernel<<<dim3(Nn / 64, Bb * Hh), 128, 0, stream>>>(
      qb, kbuf, vbuf, attb);

  gemm_bf16_kernel<1><<<dim3(M / 128, Cc / 128), 256, 0, stream>>>(
      attb, wprojb, b_proj, nullptr, nullptr, nullptr, (float*)d_out);
}
